// TinyYOLOv3_27221502722635
// MI455X (gfx1250) — compile-verified, matrix operand order checked by reading, not run on hardware
//
#include <hip/hip_runtime.h>
#include <hip/hip_bf16.h>
#include <stdint.h>

typedef __attribute__((ext_vector_type(16))) __bf16 v16bf;
typedef __attribute__((ext_vector_type(8)))  __bf16 bf8v;
typedef __attribute__((ext_vector_type(8)))  float  v8f;

#define USE_ASYNC_COPY 1
#define ROWE 40              // LDS row stride in bf16 elems (80B: 16B-aligned, bank-spread)

// ---------------------------------------------------------------------------
// Implicit-GEMM conv, 16x64 tile per wave (4 x v_wmma_f32_16x16x32_bf16).
// Block = 4 waves = 64 M-rows x 64 N-cols macro-tile.
// B (weights) staged to LDS with global_load_async_to_lds_b128, double-buffered
// and software-pipelined; fragments read back with ds_load_b128.
// Activations NHWC bf16 (channels padded to 32); weights [co][ky][kx][ci_pad],
// co padded to 64. BN folded to scale/bias; optional leaky ReLU.
// ---------------------------------------------------------------------------
__global__ __launch_bounds__(128)
void conv_wmma_kernel(const __bf16* __restrict__ in, const __bf16* __restrict__ wt,
                      const float* __restrict__ scale, const float* __restrict__ bias,
                      __bf16* __restrict__ out,
                      int N, int H, int W, int Cinp, int Cout, int Coutp,
                      int K, int pad, int leaky)
{
    __shared__ __bf16 lds[2 * 64 * ROWE];

    const int tid  = threadIdx.x;
    const int lane = tid & 31;
    const int wv   = tid >> 5;
    const int M      = N * H * W;
    const int mTiles = (M + 15) >> 4;
    int mTile = blockIdx.x * 4 + wv;
    if (mTile >= mTiles) mTile = mTiles - 1;   // keep wave alive: barriers + full EXEC
    const int nBase = blockIdx.y * 64;

    const int half = lane >> 4;
    const int lc   = lane & 15;
    const int kb   = half << 3;                // per-ISA 16-bit A/B K-slice base

    // A-fragment row (row m of the tile) -> pixel coords
    const int m   = mTile * 16 + lc;
    const bool mv = (m < M);
    const int mm  = mv ? m : 0;
    const int HW  = H * W;
    const int nimg = mm / HW;
    int rem        = mm - nimg * HW;
    const int oy   = rem / W;
    const int ox   = rem - oy * W;

    const int Kpitch = K * K * Cinp;
    const int ksteps = Kpitch >> 5;

    // ---- B-tile staging: 64 cols x 32 K bf16 = 4KB = 256 x 16B chunks ----
    auto stage = [&](int ks) {
        const int buf = ks & 1;
        #pragma unroll
        for (int r = 0; r < 2; ++r) {
            const int idx   = tid + r * 128;
            const int col   = idx >> 2;
            const int chunk = idx & 3;
            const __bf16* g = wt + (size_t)(nBase + col) * Kpitch + ks * 32 + chunk * 8;
            __bf16* l = &lds[(size_t)(buf * 64 + col) * ROWE + chunk * 8];
#if USE_ASYNC_COPY
            // generic LDS pointer: low 32 bits == LDS byte offset (aperture layout)
            unsigned loff = (unsigned)(uintptr_t)l;
            asm volatile("global_load_async_to_lds_b128 %0, %1, off"
                         :: "v"(loff), "v"((unsigned long long)(uintptr_t)g)
                         : "memory");
#else
            *(bf8v*)l = *(const bf8v*)g;
#endif
        }
    };
    auto wait_stage = [&]() {
#if USE_ASYNC_COPY
        asm volatile("s_wait_asynccnt 0" ::: "memory");
#endif
        __syncthreads();
    };

    v8f acc[4];
    #pragma unroll
    for (int s = 0; s < 4; ++s)
        #pragma unroll
        for (int i = 0; i < 8; ++i) acc[s][i] = 0.0f;

    bf8v zero8;
    #pragma unroll
    for (int i = 0; i < 8; ++i) zero8[i] = (__bf16)0.0f;

    stage(0);
    wait_stage();

    for (int ky = 0; ky < K; ++ky) {
        const int iy   = oy - pad + ky;
        const bool yok = mv && ((unsigned)iy < (unsigned)H);
        for (int kx = 0; kx < K; ++kx) {
            const int ix  = ox - pad + kx;
            const bool ok = yok && ((unsigned)ix < (unsigned)W);
            const __bf16* __restrict__ arow =
                in + ((size_t)(nimg * H + iy) * W + ix) * Cinp;
            for (int c0 = 0; c0 < Cinp; c0 += 32) {
                const int ks = ((ky * K + kx) * Cinp + c0) >> 5;
                if (ks + 1 < ksteps) stage(ks + 1);   // prefetch next B block

                union { v16bf v; bf8v h[2]; } A;
                if (ok) {
                    A.h[0] = *(const bf8v*)(arow + c0 + kb);
                    A.h[1] = *(const bf8v*)(arow + c0 + 16 + kb);
                } else {
                    A.h[0] = zero8; A.h[1] = zero8;
                }

                const __bf16* lbase = &lds[(size_t)((ks & 1) * 64) * ROWE];
                #pragma unroll
                for (int sub = 0; sub < 4; ++sub) {
                    union { v16bf v; bf8v h[2]; } B;
                    const __bf16* lb = lbase + (size_t)(sub * 16 + lc) * ROWE;
                    B.h[0] = *(const bf8v*)(lb + kb);        // ds_load_b128
                    B.h[1] = *(const bf8v*)(lb + 16 + kb);
                    acc[sub] = __builtin_amdgcn_wmma_f32_16x16x32_bf16(
                                   false, A.v, false, B.v, (short)0, acc[sub],
                                   false, false);
                }
                wait_stage();
            }
        }
    }

    // Epilogue: C/D layout -> element i: M = half*8 + i, N = lc
    #pragma unroll
    for (int sub = 0; sub < 4; ++sub) {
        const int co = nBase + sub * 16 + lc;
        if (co >= Coutp) continue;
        const bool cok = (co < Cout);
        const float s = cok ? scale[co] : 0.0f;
        const float b = cok ? bias[co]  : 0.0f;
        #pragma unroll
        for (int i = 0; i < 8; ++i) {
            const int mi = mTile * 16 + (half << 3) + i;
            if (mi < M) {
                float y = acc[sub][i] * s + b;
                if (leaky) y = (y > 0.0f) ? y : 0.1f * y;
                if (!cok)  y = 0.0f;
                out[(size_t)mi * Coutp + co] = (__bf16)y;
            }
        }
    }
}

// ---------------- input NCHW f32 -> NHWC bf16 (channel pad to 32) ----------
__global__ void nchw2nhwc_kernel(const float* __restrict__ x, __bf16* __restrict__ out,
                                 int N, int C, int H, int W, int Cp)
{
    int idx = blockIdx.x * blockDim.x + threadIdx.x;
    int total = N * H * W * Cp;
    if (idx >= total) return;
    int c = idx % Cp; int t = idx / Cp;
    int xw = t % W; t /= W;
    int y  = t % H; int n = t / H;
    float v = 0.0f;
    if (c < C) v = x[(((size_t)n * C + c) * H + y) * W + xw];
    out[idx] = (__bf16)v;
}

// ---------------- weight repack OIHW f32 -> [co][ky][kx][ci_pad] bf16 ------
__global__ void prep_w_kernel(const float* __restrict__ w, __bf16* __restrict__ wb,
                              int Cin, int Cout, int K, int Cinp, int Coutp)
{
    int idx = blockIdx.x * blockDim.x + threadIdx.x;
    int total = Coutp * K * K * Cinp;
    if (idx >= total) return;
    int ci  = idx % Cinp; int t = idx / Cinp;
    int kxy = t % (K * K); int co = t / (K * K);
    int ky = kxy / K, kx = kxy % K;
    float v = 0.0f;
    if (co < Cout && ci < Cin)
        v = w[(((size_t)co * Cin + ci) * K + ky) * K + kx];
    wb[idx] = (__bf16)v;
}

// ---------------- fold BN (or plain bias) into scale/bias ------------------
__global__ void prep_sb_kernel(const float* g, const float* b, const float* m,
                               const float* v, const float* braw,
                               float* scale, float* bias, int Cout, int Coutp, int has_bn)
{
    int co = blockIdx.x * blockDim.x + threadIdx.x;
    if (co >= Coutp) return;
    if (co >= Cout) { scale[co] = 0.0f; bias[co] = 0.0f; return; }
    if (has_bn) {
        float s = g[co] * rsqrtf(v[co] + 1e-6f);
        scale[co] = s;
        bias[co]  = b[co] - m[co] * s;
    } else {
        scale[co] = 1.0f;
        bias[co]  = braw[co];
    }
}

// ---------------- maxpool (window clipped at edges == NEG_PAD pad) ---------
__global__ void maxpool_kernel(const __bf16* __restrict__ in, __bf16* __restrict__ out,
                               int N, int Hin, int Win, int Hout, int Wout,
                               int C, int k, int s)
{
    int idx = blockIdx.x * blockDim.x + threadIdx.x;
    int total = N * Hout * Wout * C;
    if (idx >= total) return;
    int c = idx % C; int t = idx / C;
    int x = t % Wout; t /= Wout;
    int y = t % Hout; int n = t / Hout;
    float mx = -3.4e38f;
    for (int dy = 0; dy < k; ++dy) {
        int iy = y * s + dy; if (iy >= Hin) continue;
        for (int dx = 0; dx < k; ++dx) {
            int ix = x * s + dx; if (ix >= Win) continue;
            float v = (float)in[((size_t)(n * Hin + iy) * Win + ix) * C + c];
            mx = fmaxf(mx, v);
        }
    }
    out[idx] = (__bf16)mx;
}

// ---------------- 2x upsample of c11 + concat with x8 ----------------------
__global__ void upcat_kernel(const __bf16* __restrict__ up, const __bf16* __restrict__ skip,
                             __bf16* __restrict__ out, int N)
{
    int idx = blockIdx.x * blockDim.x + threadIdx.x;
    int total = N * 26 * 26 * 384;
    if (idx >= total) return;
    int c = idx % 384; int t = idx / 384;
    int x = t % 26; t /= 26;
    int y = t % 26; int n = t / 26;
    __bf16 v;
    if (c < 128)
        v = up[((size_t)(n * 13 + (y >> 1)) * 13 + (x >> 1)) * 128 + c];
    else
        v = skip[((size_t)(n * 26 + y) * 26 + x) * 256 + (c - 128)];
    out[idx] = v;
}

// ---------------- YOLO decode head -----------------------------------------
__global__ void decode_kernel(const __bf16* __restrict__ in, float* __restrict__ out,
                              int N, int G, int boxBase, int totalBoxes,
                              float aw0, float ah0, float aw1, float ah1,
                              float aw2, float ah2)
{
    int idx = blockIdx.x * blockDim.x + threadIdx.x;
    int total = N * G * G * 3;
    if (idx >= total) return;
    int b = idx % 3; int t = idx / 3;
    int x = t % G; t /= G;
    int y = t % G; int n = t / G;
    const __bf16* p = in + ((size_t)(n * G + y) * G + x) * 256 + b * 85;
    float sx = 1.0f / (1.0f + __expf(-(float)p[0]));
    float sy = 1.0f / (1.0f + __expf(-(float)p[1]));
    float bx = (sx + (float)x) / (float)G;
    float by = (sy + (float)y) / (float)G;
    float aw = (b == 0) ? aw0 : ((b == 1) ? aw1 : aw2);
    float ah = (b == 0) ? ah0 : ((b == 1) ? ah1 : ah2);
    float bw = aw * (1.0f / 416.0f) * __expf((float)p[2]);
    float bh = ah * (1.0f / 416.0f) * __expf((float)p[3]);
    float obj = 1.0f / (1.0f + __expf(-(float)p[4]));
    bool keep = obj > 1e-6f;
    float* o = out + ((size_t)n * totalBoxes + boxBase + ((y * G + x) * 3 + b)) * (size_t)85;
    o[0] = keep ? bx : 0.0f;
    o[1] = keep ? by : 0.0f;
    o[2] = keep ? bw : 0.0f;
    o[3] = keep ? bh : 0.0f;
    o[4] = keep ? obj : 0.0f;
    for (int i = 5; i < 85; ++i) {
        float s = 1.0f / (1.0f + __expf(-(float)p[i]));
        o[i] = keep ? s : 0.0f;
    }
}

// ===========================================================================
struct LSpec { int cin, cout, k, bn; };
static const LSpec LS[13] = {
    {   3,   16, 3, 1},  // c1
    {  16,   32, 3, 1},  // c2
    {  32,   64, 3, 1},  // c3
    {  64,  128, 3, 1},  // c4
    { 128,  256, 3, 1},  // c5
    { 256,  512, 3, 1},  // c6
    { 512, 1024, 3, 1},  // c7
    {1024,  256, 1, 1},  // c8
    { 256,  512, 3, 1},  // c9
    { 512,  255, 1, 0},  // c10
    { 256,  128, 1, 1},  // c11
    { 384,  256, 3, 1},  // c12
    { 256,  255, 1, 0},  // c13
};
static inline int rup32(int x) { return (x + 31) & ~31; }
static inline int rup64(int x) { return (x + 63) & ~63; }

extern "C" void kernel_launch(void* const* d_in, const int* in_sizes, int n_in,
                              void* d_out, int out_size, void* d_ws, size_t ws_size,
                              hipStream_t stream)
{
    (void)in_sizes; (void)n_in; (void)out_size; (void)ws_size;
    const int NB = 8;
    char* ws = (char*)d_ws;
    size_t off = 0;
    auto alloc = [&](size_t bytes) -> char* {
        off = (off + 255) & ~(size_t)255;
        char* p = ws + off;
        off += bytes;
        return p;
    };

    const size_t IN_ELEMS = (size_t)NB * 416 * 416 * 32;
    __bf16* IN  = (__bf16*)alloc(IN_ELEMS * 2);
    __bf16* T1  = (__bf16*)alloc(IN_ELEMS * 2);
    __bf16* X8  = (__bf16*)alloc((size_t)NB * 26 * 26 * 256 * 2);
    __bf16* X13 = (__bf16*)alloc((size_t)NB * 13 * 13 * 256 * 2);

    __bf16* wb[13]; float* scl[13]; float* bia[13];
    for (int l = 0; l < 13; ++l) {
        int Cinp = rup32(LS[l].cin), Cout64 = rup64(LS[l].cout), K = LS[l].k;
        wb[l]  = (__bf16*)alloc((size_t)Cout64 * K * K * Cinp * 2);
        scl[l] = (float*)alloc((size_t)rup32(LS[l].cout) * 4);
        bia[l] = (float*)alloc((size_t)rup32(LS[l].cout) * 4);
    }

    const float* xin = (const float*)d_in[0];
    int idx = 1;
    for (int l = 0; l < 13; ++l) {
        const float* w = (const float*)d_in[idx++];
        const float *g = nullptr, *bb = nullptr, *m = nullptr, *v = nullptr, *br = nullptr;
        if (LS[l].bn) {
            g  = (const float*)d_in[idx++];
            bb = (const float*)d_in[idx++];
            m  = (const float*)d_in[idx++];
            v  = (const float*)d_in[idx++];
        } else {
            br = (const float*)d_in[idx++];
        }
        int Cinp = rup32(LS[l].cin), Cout64 = rup64(LS[l].cout);
        int Coutp = rup32(LS[l].cout), K = LS[l].k;
        int welems = Cout64 * K * K * Cinp;
        prep_w_kernel<<<(welems + 255) / 256, 256, 0, stream>>>(
            w, wb[l], LS[l].cin, LS[l].cout, K, Cinp, Cout64);
        prep_sb_kernel<<<(Coutp + 255) / 256, 256, 0, stream>>>(
            g, bb, m, v, br, scl[l], bia[l], LS[l].cout, Coutp, LS[l].bn);
    }

    auto conv = [&](int l, const __bf16* inp, __bf16* outp, int H, int W) {
        int Cinp = rup32(LS[l].cin), Coutp = rup32(LS[l].cout);
        int Cout64 = rup64(LS[l].cout);
        int M = NB * H * W;
        int mTiles = (M + 15) / 16;
        dim3 grid((mTiles + 3) / 4, Cout64 / 64);
        int leaky = (l == 9 || l == 12) ? 0 : 1;   // c10, c13: no activation
        conv_wmma_kernel<<<grid, 128, 0, stream>>>(
            inp, wb[l], scl[l], bia[l], outp,
            NB, H, W, Cinp, LS[l].cout, Coutp, LS[l].k, (LS[l].k == 3) ? 1 : 0, leaky);
    };
    auto pool = [&](const __bf16* inp, __bf16* outp, int Hin, int Hout, int C, int k, int s) {
        int total = NB * Hout * Hout * C;
        maxpool_kernel<<<(total + 255) / 256, 256, 0, stream>>>(
            inp, outp, NB, Hin, Hin, Hout, Hout, C, k, s);
    };

    {
        int total = NB * 416 * 416 * 32;
        nchw2nhwc_kernel<<<(total + 255) / 256, 256, 0, stream>>>(
            xin, IN, NB, 3, 416, 416, 32);
    }
    conv(0, IN, T1, 416, 416);          // c1
    pool(T1, IN, 416, 208, 32, 2, 2);
    conv(1, IN, T1, 208, 208);          // c2
    pool(T1, IN, 208, 104, 32, 2, 2);
    conv(2, IN, T1, 104, 104);          // c3
    pool(T1, IN, 104, 52, 64, 2, 2);
    conv(3, IN, T1, 52, 52);            // c4
    pool(T1, IN, 52, 26, 128, 2, 2);
    conv(4, IN, X8, 26, 26);            // c5 -> x8
    pool(X8, IN, 26, 13, 256, 2, 2);
    conv(5, IN, T1, 13, 13);            // c6
    pool(T1, IN, 13, 13, 512, 2, 1);    // stride-1 pool (clipped window)
    conv(6, IN, T1, 13, 13);            // c7
    conv(7, T1, X13, 13, 13);           // c8 -> x13
    conv(8, X13, IN, 13, 13);           // c9
    conv(9, IN, T1, 13, 13);            // c10 (no act)

    {   // decode y1 (13x13 head) -> boxes [2028, 2535)
        int total = NB * 13 * 13 * 3;
        decode_kernel<<<(total + 255) / 256, 256, 0, stream>>>(
            T1, (float*)d_out, NB, 13, 2028, 2535,
            81.0f, 82.0f, 135.0f, 169.0f, 344.0f, 319.0f);
    }

    conv(10, X13, IN, 13, 13);          // c11
    {
        int total = NB * 26 * 26 * 384;
        upcat_kernel<<<(total + 255) / 256, 256, 0, stream>>>(IN, X8, T1, NB);
    }
    conv(11, T1, IN, 26, 26);           // c12
    conv(12, IN, T1, 26, 26);           // c13 (no act)

    {   // decode y2 (26x26 head) -> boxes [0, 2028)
        int total = NB * 26 * 26 * 3;
        decode_kernel<<<(total + 255) / 256, 256, 0, stream>>>(
            T1, (float*)d_out, NB, 26, 0, 2535,
            10.0f, 14.0f, 23.0f, 27.0f, 37.0f, 58.0f);
    }
}
